// WindowAttention_49589692400259
// MI455X (gfx1250) — matrix operand order repaired and checked by reading, not compiled, not run on hardware
//
#include <hip/hip_runtime.h>
#include <hip/hip_bf16.h>

typedef _Float16 v4h  __attribute__((ext_vector_type(4)));
typedef _Float16 v8h  __attribute__((ext_vector_type(8)));
typedef _Float16 v16h __attribute__((ext_vector_type(16)));
typedef float    v8f  __attribute__((ext_vector_type(8)));

static __device__ __forceinline__ v16h cat16(v8h lo, v8h hi) {
  return __builtin_shufflevector(lo, hi, 0,1,2,3,4,5,6,7,8,9,10,11,12,13,14,15);
}
// A-matrix fragment (16x32 f16): lane (g,m): elems 0-7 = k[g*8 .. g*8+7],
// elems 8-15 = k[16+g*8 .. 16+g*8+7]  -> two b128 LDS loads at p and p+16.
static __device__ __forceinline__ v16h lds_a16(const _Float16* p) {
  return cat16(*(const v8h*)p, *(const v8h*)(p + 16));
}
// B-matrix fragment (32x16 f16): lane (g,n): elems 0-7 = k[g*8 .. g*8+7],
// elems 8-15 = k[16+g*8 .. 16+g*8+7]  -> two b128 LDS loads at p and p+16.
static __device__ __forceinline__ v16h lds_b16(const _Float16* p) {
  return cat16(*(const v8h*)p, *(const v8h*)(p + 16));
}

#define QS_S 40    // 32 + 8 pad (halves)
#define VT_S 264   // 256 + 8 pad (halves)
#define P_S  264

// One workgroup per (b, h). 256 threads = 8 waves; each wave handles two
// 16-row q blocks. Scores for a full 16x256 strip live in 16 WMMA accumulators.
__global__ __launch_bounds__(256) void swin_attn_kernel(
    const float* __restrict__ qkv,
    const int*   __restrict__ rpi,
    const float* __restrict__ mask,
    const float* __restrict__ rpb,
    float* __restrict__ xout)
{
  extern __shared__ _Float16 smem[];
  _Float16* Qs = smem;                 // [256][QS_S]
  _Float16* Ks = Qs + 256 * QS_S;      // [256][QS_S]
  _Float16* Vt = Ks + 256 * QS_S;      // [32][VT_S]  (transposed V: hd-major)
  _Float16* Pb = Vt + 32  * VT_S;      // [8 waves][16][P_S]

  const int b    = blockIdx.x / 6;
  const int h    = blockIdx.x % 6;
  const int tid  = threadIdx.x;
  const int lane = tid & 31;
  const int wave = tid >> 5;
  const int ll   = lane & 15;
  const int g    = lane >> 4;
  const float scale = 0.17677669529663687f;  // 1/sqrt(32)

  // ---- stage Q (pre-scaled), K, V^T into LDS as f16 ----
  {
    const float* base = qkv + (size_t)(b * 256) * 576 + h * 32;
    const int rsub = tid >> 3;         // 0..31
    const int d4   = (tid & 7) << 2;   // 0..28
    for (int p = 0; p < 8; ++p) {
      const int n = (p << 5) + rsub;
      const float* rp = base + n * 576 + d4;
      float4 q4 = *(const float4*)(rp);
      float4 k4 = *(const float4*)(rp + 192);
      float4 v4 = *(const float4*)(rp + 384);
      v4h qh = { (_Float16)(q4.x * scale), (_Float16)(q4.y * scale),
                 (_Float16)(q4.z * scale), (_Float16)(q4.w * scale) };
      v4h kh = { (_Float16)k4.x, (_Float16)k4.y, (_Float16)k4.z, (_Float16)k4.w };
      *(v4h*)(Qs + n * QS_S + d4) = qh;
      *(v4h*)(Ks + n * QS_S + d4) = kh;
      Vt[(d4 + 0) * VT_S + n] = (_Float16)v4.x;
      Vt[(d4 + 1) * VT_S + n] = (_Float16)v4.y;
      Vt[(d4 + 2) * VT_S + n] = (_Float16)v4.z;
      Vt[(d4 + 3) * VT_S + n] = (_Float16)v4.w;
    }
  }
  __syncthreads();

  _Float16* Pw = Pb + wave * (16 * P_S);
  const float* mbase = mask + (size_t)(b & 63) * 65536;

  for (int qb = wave; qb < 16; qb += 8) {
    const int q0 = qb << 4;
    const v16h aQ = lds_a16(Qs + (q0 + ll) * QS_S + g * 8);

    // ---- S = Q * K^T : 16 tiles of 16x16 covering all 256 keys ----
    v8f acc[16];
    #pragma unroll
    for (int j = 0; j < 16; ++j) {
      v16h bK = lds_b16(Ks + ((j << 4) + ll) * QS_S + g * 8);
      v8f z = {};
      acc[j] = __builtin_amdgcn_wmma_f32_16x16x32_f16(
          false, aQ, false, bK, (short)0, z, false, false);
    }

    // ---- + relative-position bias + window mask, row max ----
    const int qrow = q0 + (g << 3);
    float rmax[8];
    #pragma unroll
    for (int r = 0; r < 8; ++r) rmax[r] = -3.0e38f;
    #pragma unroll
    for (int j = 0; j < 16; ++j) {
      const int key = (j << 4) + ll;
      #pragma unroll
      for (int r = 0; r < 8; ++r) {
        const int q = qrow + r;
        const float bv = rpb[rpi[q * 256 + key] * 6 + h];
        float s = acc[j][r] + bv + mbase[q * 256 + key];
        acc[j][r] = s;
        rmax[r] = fmaxf(rmax[r], s);
      }
    }
    #pragma unroll
    for (int r = 0; r < 8; ++r) {
      float m = rmax[r];
      m = fmaxf(m, __shfl_xor(m, 1, 32));
      m = fmaxf(m, __shfl_xor(m, 2, 32));
      m = fmaxf(m, __shfl_xor(m, 4, 32));
      m = fmaxf(m, __shfl_xor(m, 8, 32));
      rmax[r] = m;
    }

    // ---- exp + row sum ----
    float rsum[8];
    #pragma unroll
    for (int r = 0; r < 8; ++r) rsum[r] = 0.0f;
    #pragma unroll
    for (int j = 0; j < 16; ++j) {
      #pragma unroll
      for (int r = 0; r < 8; ++r) {
        float e = __expf(acc[j][r] - rmax[r]);
        acc[j][r] = e;
        rsum[r] += e;
      }
    }
    #pragma unroll
    for (int r = 0; r < 8; ++r) {
      float s = rsum[r];
      s += __shfl_xor(s, 1, 32);
      s += __shfl_xor(s, 2, 32);
      s += __shfl_xor(s, 4, 32);
      s += __shfl_xor(s, 8, 32);
      rsum[r] = 1.0f / s;
    }

    // ---- normalized P -> LDS (f16, row-major, padded) ----
    #pragma unroll
    for (int j = 0; j < 16; ++j) {
      #pragma unroll
      for (int r = 0; r < 8; ++r) {
        Pw[((g << 3) + r) * P_S + (j << 4) + ll] =
            (_Float16)(acc[j][r] * rsum[r]);
      }
    }
    asm volatile("s_wait_dscnt 0" ::: "memory");  // within-wave LDS RAW

    // ---- O = P * V ----
    #pragma unroll
    for (int db = 0; db < 2; ++db) {
      v8f o = {};
      #pragma unroll
      for (int kb = 0; kb < 8; ++kb) {
        v16h aP = lds_a16(Pw + ll * P_S + kb * 32 + g * 8);
        v16h bV = lds_b16(Vt + (db * 16 + ll) * VT_S + kb * 32 + g * 8);
        o = __builtin_amdgcn_wmma_f32_16x16x32_f16(
            false, aP, false, bV, (short)0, o, false, false);
      }
      float* op = xout + ((size_t)(b * 256) + q0 + (g << 3)) * 192
                       + h * 32 + db * 16 + ll;
      #pragma unroll
      for (int r = 0; r < 8; ++r) op[r * 192] = o[r];
    }
  }
}

// In-place projection: out = x @ W^T + bias. Each workgroup owns 64 rows of
// one batch; it reads its tile + W into LDS before the barrier, then
// overwrites the same rows (workgroup-private, so in-place is safe).
__global__ __launch_bounds__(256) void swin_proj_kernel(
    const float* __restrict__ pw, const float* __restrict__ pb,
    float* __restrict__ xo)
{
  extern __shared__ _Float16 smem[];
  _Float16* Ws = smem;             // [192][200]  W[c'][c], k-contiguous
  _Float16* Xs = Ws + 192 * 200;   // [64][200]
  const int b    = blockIdx.x;
  const int rt   = blockIdx.y;     // 64-row tile
  const int tid  = threadIdx.x;
  const int lane = tid & 31;
  const int wave = tid >> 5;
  const int ll   = lane & 15;
  const int g    = lane >> 4;

  for (int i = tid; i < 9216; i += 256) {          // 192*192 floats / 4
    const int e = i << 2;
    float4 w4 = *(const float4*)(pw + e);
    const int row = e / 192, col = e % 192;
    v4h wh = { (_Float16)w4.x, (_Float16)w4.y, (_Float16)w4.z, (_Float16)w4.w };
    *(v4h*)(Ws + row * 200 + col) = wh;
  }
  float* xbase = xo + ((size_t)(b * 256) + rt * 64) * 192;
  for (int i = tid; i < 3072; i += 256) {          // 64*192 floats / 4
    const int e = i << 2;
    float4 x4 = *(const float4*)(xbase + e);
    const int row = e / 192, col = e % 192;
    v4h xh = { (_Float16)x4.x, (_Float16)x4.y, (_Float16)x4.z, (_Float16)x4.w };
    *(v4h*)(Xs + row * 200 + col) = xh;
  }
  __syncthreads();

  // 48 output tiles (4 row-blocks x 12 col-blocks); 6 per wave.
  for (int t = wave * 6; t < wave * 6 + 6; ++t) {
    const int rb = t / 12, cb = t % 12;
    v8f o = {};
    #pragma unroll
    for (int kb = 0; kb < 6; ++kb) {
      v16h a = lds_a16(Xs + (rb * 16 + ll) * 200 + kb * 32 + g * 8);
      v16h w = lds_b16(Ws + (cb * 16 + ll) * 200 + kb * 32 + g * 8);
      o = __builtin_amdgcn_wmma_f32_16x16x32_f16(
          false, a, false, w, (short)0, o, false, false);
    }
    const int c = cb * 16 + ll;
    const float bv = pb[c];
    float* op = xbase + (rb * 16 + (g << 3)) * 192 + c;
    #pragma unroll
    for (int r = 0; r < 8; ++r) op[r * 192] = o[r] + bv;
  }
}

extern "C" void kernel_launch(void* const* d_in, const int* in_sizes, int n_in,
                              void* d_out, int out_size, void* d_ws, size_t ws_size,
                              hipStream_t stream) {
  const float* qkv  = (const float*)d_in[0];
  const int*   rpi  = (const int*)d_in[1];
  const float* mask = (const float*)d_in[2];
  const float* rpb  = (const float*)d_in[3];
  const float* pw   = (const float*)d_in[4];
  const float* pb   = (const float*)d_in[5];
  float* out = (float*)d_out;

  const size_t lds_attn =
      (size_t)(256 * QS_S + 256 * QS_S + 32 * VT_S + 8 * 16 * P_S) * 2;  // f16
  swin_attn_kernel<<<dim3(512 * 6), dim3(256), lds_attn, stream>>>(
      qkv, rpi, mask, rpb, out);

  const size_t lds_proj = (size_t)(192 * 200 + 64 * 200) * 2;
  swin_proj_kernel<<<dim3(512, 4), dim3(256), lds_proj, stream>>>(pw, pb, out);
}
